// Spito_inter_44487271252007
// MI455X (gfx1250) — compile-verified, matrix operand order checked by reading, not run on hardware
//
#include <hip/hip_runtime.h>
#include <hip/hip_bf16.h>
#include <math.h>

typedef _Float16 half_t;
typedef __attribute__((ext_vector_type(16))) _Float16 v16h;
typedef __attribute__((ext_vector_type(8)))  _Float16 v8h;
typedef __attribute__((ext_vector_type(8)))  float    v8f;

#define HH       128   // H
#define DD       3     // D
#define NFD      3     // NF
#define FE       7     // F_EDGE = 2*NF + EF
#define NET_IN   305
#define NET_K    320   // padded K for edge MLP layer 1
#define NET_OUT  149
#define SELF_IN  292
#define SELF_K   320
#define SELF_OUT 146
#define NTO      10    // 160 output cols (padded)
#define ROWS     32    // M rows (edges/nodes) per wave = two 16-row WMMA tiles

__device__ __forceinline__ float siluf(float x) {
    return x * (1.0f / (1.0f + __expf(-x)));
}

// Load a WMMA 16-bit fragment: halves [0..7] = K kb..kb+7, [8..15] = K kb+16..kb+23
__device__ __forceinline__ v16h load_frag(const half_t* p) {
    v8h lo = *(const v8h*)(p);
    v8h hi = *(const v8h*)(p + 16);
    v16h r;
#pragma unroll
    for (int i = 0; i < 8; ++i) { r[i] = lo[i]; r[i + 8] = hi[i]; }
    return r;
}

// C[2 M-tiles] = A(32 x KT*32, row stride a_stride halves) * W^T + bias
// W stored transposed+padded: wt[n][KT*32] f16. Each B fragment is reused for
// both M tiles (halves weight traffic per WMMA).
template<int NT, int KT>
__device__ __forceinline__ void mma_block2(const half_t* a_base, int a_stride,
                                           const half_t* __restrict__ wt,
                                           const float* __restrict__ bias, int bias_n,
                                           int ln, int g, v8f (&acc)[2][NTO]) {
#pragma unroll
    for (int t = 0; t < NT; ++t) {
        int n = t * 16 + ln;
        float bv = (n < bias_n) ? bias[n] : 0.0f;
#pragma unroll
        for (int ms = 0; ms < 2; ++ms)
#pragma unroll
            for (int i = 0; i < 8; ++i) acc[ms][t][i] = bv;
    }
    for (int kt = 0; kt < KT; ++kt) {
        const half_t* ap = a_base + ln * a_stride + kt * 32 + g * 8;
        v16h a0 = load_frag(ap);
        v16h a1 = load_frag(ap + 16 * a_stride);
#pragma unroll
        for (int t = 0; t < NT; ++t) {
            v16h bf = load_frag(wt + (size_t)(t * 16 + ln) * (KT * 32) + kt * 32 + g * 8);
            acc[0][t] = __builtin_amdgcn_wmma_f32_16x16x32_f16(
                false, a0, false, bf, (short)0, acc[0][t], false, false);
            acc[1][t] = __builtin_amdgcn_wmma_f32_16x16x32_f16(
                false, a1, false, bf, (short)0, acc[1][t], false, false);
        }
    }
}

// ---------------- weight prep: f32 W[K][N] -> f16 Wt[nPad][kPad] ----------------
__global__ void prep_w_kernel(const float* __restrict__ W, half_t* __restrict__ Wt,
                              int K, int N, int Kpad, int Npad) {
    int idx = blockIdx.x * blockDim.x + threadIdx.x;
    int total = Kpad * Npad;
    if (idx >= total) return;
    int n = idx / Kpad;
    int k = idx - n * Kpad;
    float v = (n < N && k < K) ? W[(size_t)k * N + n] : 0.0f;
    Wt[idx] = (half_t)v;
}

__global__ void zero_kernel(float* __restrict__ p, long n) {
    long i = (long)blockIdx.x * blockDim.x + threadIdx.x;
    long stride = (long)gridDim.x * blockDim.x;
    for (; i < n; i += stride) p[i] = 0.0f;
}

__global__ void count_kernel(const int* __restrict__ ei, float* __restrict__ cnt, int E) {
    int i = blockIdx.x * blockDim.x + threadIdx.x;
    if (i < E) atomicAdd(&cnt[ei[i]], 1.0f);
}

// ---------------- edge stage: 32 edges per wave, 1 wave per workgroup ----------------
__global__ __launch_bounds__(32)
void edge_kernel(const float* __restrict__ fcur, const float* __restrict__ scur,
                 const int* __restrict__ ei, const float* __restrict__ ef,
                 const half_t* __restrict__ w1t, const half_t* __restrict__ w2t,
                 const half_t* __restrict__ w3t,
                 const float* __restrict__ b1, const float* __restrict__ b2,
                 const float* __restrict__ b3,
                 float* __restrict__ facc, float* __restrict__ sacc,
                 int E) {
    __shared__ half_t xls[ROWS][NET_K];
    __shared__ half_t hls[ROWS][HH];
    __shared__ float  fls[ROWS][DD * FE];
    __shared__ float  ols[ROWS][FE * NFD];
    __shared__ int    esrc[ROWS];

    const int lane = threadIdx.x;
    const int ln = lane & 15, g = lane >> 4;
    const int e0 = blockIdx.x * ROWS;

    // stage 1 (each lane = one edge): build _f, gram f2s (49) + l2norm, pad tail
    {
        int m = lane;
        int e = e0 + m;
        int ec = (e < E) ? e : (E - 1);
        int src = ei[ec];
        int dst = ei[E + ec];
        esrc[m] = (e < E) ? src : -1;
        for (int d = 0; d < DD; ++d) {
            for (int i = 0; i < NFD; ++i) {
                fls[m][d * FE + i]       = fcur[(size_t)src * (DD * NFD) + d * NFD + i];
                fls[m][d * FE + NFD + i] = fcur[(size_t)dst * (DD * NFD) + d * NFD + i];
            }
            fls[m][d * FE + 2 * NFD] = ef[(size_t)ec * DD + d];
        }
        float ss = 0.0f;
        for (int i = 0; i < FE; ++i)
            for (int j = 0; j < FE; ++j) {
                float gv = 0.0f;
                for (int d = 0; d < DD; ++d)
                    gv += fls[m][d * FE + i] * fls[m][d * FE + j];
                ss += gv * gv;
            }
        float inv = 1.0f / fmaxf(sqrtf(ss), 1e-12f);
        for (int i = 0; i < FE; ++i)
            for (int j = 0; j < FE; ++j) {
                float gv = 0.0f;
                for (int d = 0; d < DD; ++d)
                    gv += fls[m][d * FE + i] * fls[m][d * FE + j];
                xls[m][i * FE + j] = (half_t)(gv * inv);
            }
        for (int k = NET_IN; k < NET_K; ++k) xls[m][k] = (half_t)0.0f;
    }
    // stage 2: gather s[src] (cols 49..176) and s[dst] (cols 177..304); 64 rows over 2 passes
#pragma unroll
    for (int rep = 0; rep < 2; ++rep) {
        int m = lane;          // edge row
        int hsel = rep;        // 0 = src block, 1 = dst block
        int e = e0 + m;
        int ec = (e < E) ? e : (E - 1);
        int node = hsel ? ei[E + ec] : ei[ec];
        int base = FE * FE + hsel * HH;
        const float* sp = scur + (size_t)node * HH;
        for (int k = 0; k < HH; k += 4) {
            float4 v = *(const float4*)(sp + k);
            xls[m][base + k + 0] = (half_t)v.x;
            xls[m][base + k + 1] = (half_t)v.y;
            xls[m][base + k + 2] = (half_t)v.z;
            xls[m][base + k + 3] = (half_t)v.w;
        }
    }
    __syncthreads();

    v8f acc[2][NTO];
    // L1: 305(pad320) -> 128, silu
    mma_block2<8, NET_K / 32>(&xls[0][0], NET_K, w1t, b1, HH, ln, g, acc);
#pragma unroll
    for (int t = 0; t < 8; ++t) {
        int n = t * 16 + ln;
#pragma unroll
        for (int ms = 0; ms < 2; ++ms)
#pragma unroll
            for (int v = 0; v < 8; ++v)
                hls[ms * 16 + v + 8 * g][n] = (half_t)siluf(acc[ms][t][v]);
    }
    __syncthreads();
    // L2: 128 -> 128, silu (result reuses xls rows, stride NET_K)
    mma_block2<8, HH / 32>(&hls[0][0], HH, w2t, b2, HH, ln, g, acc);
#pragma unroll
    for (int t = 0; t < 8; ++t) {
        int n = t * 16 + ln;
#pragma unroll
        for (int ms = 0; ms < 2; ++ms)
#pragma unroll
            for (int v = 0; v < 8; ++v)
                xls[ms * 16 + v + 8 * g][n] = (half_t)siluf(acc[ms][t][v]);
    }
    __syncthreads();
    // L3: 128 -> 149(pad160)
    mma_block2<NTO, HH / 32>(&xls[0][0], NET_K, w3t, b3, NET_OUT, ln, g, acc);

    // split: cols [0,21) -> f_c staging; cols [21,149) -> scatter-add s_acc[src]
#pragma unroll
    for (int t = 0; t < NTO; ++t) {
        int n = t * 16 + ln;
#pragma unroll
        for (int ms = 0; ms < 2; ++ms)
#pragma unroll
            for (int v = 0; v < 8; ++v) {
                int m = ms * 16 + v + 8 * g;
                float val = acc[ms][t][v];
                if (n < FE * NFD) {
                    ols[m][n] = val;
                } else if (n < NET_OUT) {
                    int src = esrc[m];
                    if (src >= 0) atomicAdd(&sacc[(size_t)src * HH + (n - FE * NFD)], val);
                }
            }
    }
    __syncthreads();
    // f_c[e] = _f (3x7) @ c_f (7x3), scatter-add to f_acc[src]
    {
        int m = lane;
        int src = esrc[m];
        if (src >= 0) {
            for (int d = 0; d < DD; ++d)
                for (int kk = 0; kk < NFD; ++kk) {
                    float a = 0.0f;
                    for (int j = 0; j < FE; ++j)
                        a += fls[m][d * FE + j] * ols[m][j * NFD + kk];
                    atomicAdd(&facc[(size_t)src * (DD * NFD) + d * NFD + kk], a);
                }
        }
    }
}

// ---------------- node stage: 32 nodes per wave ----------------
__global__ __launch_bounds__(32)
void node_kernel(const float* __restrict__ fcur, const float* __restrict__ scur,
                 const float* __restrict__ facc, const float* __restrict__ sacc,
                 const float* __restrict__ cnt,
                 const half_t* __restrict__ w1t, const half_t* __restrict__ w2t,
                 const half_t* __restrict__ w3t,
                 const float* __restrict__ b1, const float* __restrict__ b2,
                 const float* __restrict__ b3,
                 float* __restrict__ fnext, float* __restrict__ snext,
                 int Nn) {
    __shared__ half_t xls[ROWS][SELF_K];
    __shared__ half_t hls[ROWS][HH];
    __shared__ float  tfs[ROWS][DD * 2 * NFD];   // 32 x (3 x 6)
    __shared__ float  ols[ROWS][2 * NFD * NFD];  // 32 x 18

    const int lane = threadIdx.x;
    const int ln = lane & 15, g = lane >> 4;
    const int n0 = blockIdx.x * ROWS;
    const int TF = 2 * NFD;  // 6

    {
        int m = lane;
        int node = n0 + m; if (node >= Nn) node = Nn - 1;
        float inv = 1.0f / fmaxf(cnt[node], 1.0f);
        for (int d = 0; d < DD; ++d)
            for (int i = 0; i < NFD; ++i) {
                tfs[m][d * TF + i]       = fcur[(size_t)node * (DD * NFD) + d * NFD + i];
                tfs[m][d * TF + NFD + i] = facc[(size_t)node * (DD * NFD) + d * NFD + i] * inv;
            }
        float ss = 0.0f;
        for (int i = 0; i < TF; ++i)
            for (int j = 0; j < TF; ++j) {
                float gv = 0.0f;
                for (int d = 0; d < DD; ++d)
                    gv += tfs[m][d * TF + i] * tfs[m][d * TF + j];
                ss += gv * gv;
            }
        float inv2 = 1.0f / fmaxf(sqrtf(ss), 1e-12f);
        for (int i = 0; i < TF; ++i)
            for (int j = 0; j < TF; ++j) {
                float gv = 0.0f;
                for (int d = 0; d < DD; ++d)
                    gv += tfs[m][d * TF + i] * tfs[m][d * TF + j];
                xls[m][i * TF + j] = (half_t)(gv * inv2);
            }
        for (int k = SELF_IN; k < SELF_K; ++k) xls[m][k] = (half_t)0.0f;
    }
    // gather s (cols 36..163) and s_c = s_acc/cnt (cols 164..291)
#pragma unroll
    for (int rep = 0; rep < 2; ++rep) {
        int m = lane;
        int node = n0 + m; if (node >= Nn) node = Nn - 1;
        float scale = rep ? (1.0f / fmaxf(cnt[node], 1.0f)) : 1.0f;
        const float* sp = (rep ? sacc : scur) + (size_t)node * HH;
        int base = TF * TF + rep * HH;
        for (int k = 0; k < HH; k += 4) {
            float4 v = *(const float4*)(sp + k);
            xls[m][base + k + 0] = (half_t)(v.x * scale);
            xls[m][base + k + 1] = (half_t)(v.y * scale);
            xls[m][base + k + 2] = (half_t)(v.z * scale);
            xls[m][base + k + 3] = (half_t)(v.w * scale);
        }
    }
    __syncthreads();

    v8f acc[2][NTO];
    mma_block2<8, SELF_K / 32>(&xls[0][0], SELF_K, w1t, b1, HH, ln, g, acc);
#pragma unroll
    for (int t = 0; t < 8; ++t) {
        int n = t * 16 + ln;
#pragma unroll
        for (int ms = 0; ms < 2; ++ms)
#pragma unroll
            for (int v = 0; v < 8; ++v)
                hls[ms * 16 + v + 8 * g][n] = (half_t)siluf(acc[ms][t][v]);
    }
    __syncthreads();
    mma_block2<8, HH / 32>(&hls[0][0], HH, w2t, b2, HH, ln, g, acc);
#pragma unroll
    for (int t = 0; t < 8; ++t) {
        int n = t * 16 + ln;
#pragma unroll
        for (int ms = 0; ms < 2; ++ms)
#pragma unroll
            for (int v = 0; v < 8; ++v)
                xls[ms * 16 + v + 8 * g][n] = (half_t)siluf(acc[ms][t][v]);
    }
    __syncthreads();
    mma_block2<NTO, HH / 32>(&xls[0][0], SELF_K, w3t, b3, SELF_OUT, ln, g, acc);

    // split: cols [0,18) -> temp_f_c staging; cols [18,146) -> s_next
#pragma unroll
    for (int t = 0; t < NTO; ++t) {
        int n = t * 16 + ln;
#pragma unroll
        for (int ms = 0; ms < 2; ++ms)
#pragma unroll
            for (int v = 0; v < 8; ++v) {
                int m = ms * 16 + v + 8 * g;
                int node = n0 + m;
                float val = acc[ms][t][v];
                if (n < TF * NFD) {
                    ols[m][n] = val;
                } else if (n < SELF_OUT) {
                    if (node < Nn) snext[(size_t)node * HH + (n - TF * NFD)] = val;
                }
            }
    }
    __syncthreads();
    {
        int m = lane;
        int node = n0 + m;
        if (node < Nn) {
            for (int d = 0; d < DD; ++d)
                for (int kk = 0; kk < NFD; ++kk) {
                    float a = 0.0f;
                    for (int j = 0; j < TF; ++j)
                        a += tfs[m][d * TF + j] * ols[m][j * NFD + kk];
                    fnext[(size_t)node * (DD * NFD) + d * NFD + kk] = a;
                }
        }
    }
}

extern "C" void kernel_launch(void* const* d_in, const int* in_sizes, int n_in,
                              void* d_out, int out_size, void* d_ws, size_t ws_size,
                              hipStream_t stream) {
    const float* f_in  = (const float*)d_in[0];
    const float* s_in  = (const float*)d_in[1];
    const int*   ei    = (const int*)d_in[2];
    const float* ef    = (const float*)d_in[3];
    const float* nw1 = (const float*)d_in[4];  const float* nb1 = (const float*)d_in[5];
    const float* nw2 = (const float*)d_in[6];  const float* nb2 = (const float*)d_in[7];
    const float* nw3 = (const float*)d_in[8];  const float* nb3 = (const float*)d_in[9];
    const float* sw1 = (const float*)d_in[10]; const float* sb1 = (const float*)d_in[11];
    const float* sw2 = (const float*)d_in[12]; const float* sb2 = (const float*)d_in[13];
    const float* sw3 = (const float*)d_in[14]; const float* sb3 = (const float*)d_in[15];

    const int Nn = in_sizes[0] / (DD * NFD);
    const int E  = in_sizes[2] / 2;

    // workspace layout
    char* ws = (char*)d_ws;
    size_t off = 0;
    auto alloc_f = [&](size_t n) { float* p = (float*)(ws + off); off += n * sizeof(float); return p; };
    float* fA   = alloc_f((size_t)Nn * 9);
    float* fB   = alloc_f((size_t)Nn * 9);
    float* sA   = alloc_f((size_t)Nn * HH);
    float* sB   = alloc_f((size_t)Nn * HH);
    float* facc = alloc_f((size_t)Nn * 9);
    float* sacc = alloc_f((size_t)Nn * HH);
    float* cnt  = alloc_f((size_t)Nn);
    off = (off + 255) & ~(size_t)255;
    auto alloc_h = [&](size_t n) { half_t* p = (half_t*)(ws + off); off += n * sizeof(half_t); return p; };
    half_t* nW1t = alloc_h((size_t)128 * NET_K);
    half_t* nW2t = alloc_h((size_t)128 * HH);
    half_t* nW3t = alloc_h((size_t)160 * HH);
    half_t* sW1t = alloc_h((size_t)128 * SELF_K);
    half_t* sW2t = alloc_h((size_t)128 * HH);
    half_t* sW3t = alloc_h((size_t)160 * HH);
    (void)ws_size; (void)n_in; (void)out_size;

    // weight prep (transpose + pad + f16)
    auto prep = [&](const float* W, half_t* Wt, int K, int N, int Kp, int Np) {
        int tot = Kp * Np;
        prep_w_kernel<<<(tot + 255) / 256, 256, 0, stream>>>(W, Wt, K, N, Kp, Np);
    };
    prep(nw1, nW1t, NET_IN, HH, NET_K, 128);
    prep(nw2, nW2t, HH, HH, HH, 128);
    prep(nw3, nW3t, HH, NET_OUT, HH, 160);
    prep(sw1, sW1t, SELF_IN, HH, SELF_K, 128);
    prep(sw2, sW2t, HH, HH, HH, 128);
    prep(sw3, sW3t, HH, SELF_OUT, HH, 160);

    hipMemcpyAsync(fA, f_in, (size_t)Nn * 9 * sizeof(float), hipMemcpyDeviceToDevice, stream);
    hipMemcpyAsync(sA, s_in, (size_t)Nn * HH * sizeof(float), hipMemcpyDeviceToDevice, stream);

    // segment counts (same every layer)
    zero_kernel<<<256, 256, 0, stream>>>(cnt, (long)Nn);
    count_kernel<<<(E + 255) / 256, 256, 0, stream>>>(ei, cnt, E);

    const int blkE = (E + ROWS - 1) / ROWS;
    const int blkN = (Nn + ROWS - 1) / ROWS;

    float* fc = fA; float* fn = fB; float* sc = sA; float* sn = sB;
    for (int step = 0; step < 4; ++step) {
        zero_kernel<<<256, 256, 0, stream>>>(facc, (long)Nn * 9);
        zero_kernel<<<256, 256, 0, stream>>>(sacc, (long)Nn * HH);
        edge_kernel<<<blkE, 32, 0, stream>>>(fc, sc, ei, ef,
            nW1t, nW2t, nW3t, nb1, nb2, nb3, facc, sacc, E);
        node_kernel<<<blkN, 32, 0, stream>>>(fc, sc, facc, sacc, cnt,
            sW1t, sW2t, sW3t, sb1, sb2, sb3, fn, sn, Nn);
        float* t;
        t = fc; fc = fn; fn = t;
        t = sc; sc = sn; sn = t;
    }

    float* out = (float*)d_out;
    hipMemcpyAsync(out, fc, (size_t)Nn * 9 * sizeof(float), hipMemcpyDeviceToDevice, stream);
    hipMemcpyAsync(out + (size_t)Nn * 9, sc, (size_t)Nn * HH * sizeof(float), hipMemcpyDeviceToDevice, stream);
}
